// CrossViewEncoder_59476707115286
// MI455X (gfx1250) — matrix operand order repaired and checked by reading, not compiled, not run on hardware
//
#include <hip/hip_runtime.h>
#include <hip/hip_bf16.h>

// ---------------------------------------------------------------------------
// CrossViewEncoder for MI455X (gfx1250, wave32).
// Dense GEMMs -> v_wmma_f32_16x16x32_bf16, one wave = 16 rows x 128 cols
// (8 accumulators), software-pipelined: B fragments double-buffered in groups
// of 4 so >=8 b128 loads are in flight across every WMMA group (partial
// loadcnt waits instead of wait-0). K is a template constant so all fragment
// addresses fold into immediate offsets. Weights pre-transposed to bf16
// [dout][din]; fragments are 16B b128 loads matching CDNA5 A/B layouts.
// ---------------------------------------------------------------------------

#define NN 50000
#define EE 400000
#define DD 128
#define HH 8
#define TT 20
#define MM 6
#define FFD 512

typedef __attribute__((ext_vector_type(16))) __bf16 v16bf;
typedef __attribute__((ext_vector_type(8)))  float  v8f;
typedef __attribute__((ext_vector_type(4)))  unsigned int u4v;
typedef __attribute__((ext_vector_type(4)))  float  f4v;

union FragU { v16bf v; u4v q[2]; };

// GEMM epilogue flags
#define GF_BIAS  1
#define GF_RELU  2
#define GF_ADD   4
#define GF_BF16  8
#define GF_MODES 16

// ---------------------------------------------------------------------------
// WMMA GEMM: Y[rows,Kout] = act(X[rows,K] @ W[K,Kout] + bias)
// X bf16 row-major, Wt bf16 TRANSPOSED [Kout][K]. rows%16==0, K%32==0,
// Kout%128==0. Fragment layouts per CDNA5 ISA:
//   A 16x32: lanes 0-15 row=lane, halves 0..7 -> K+0..7, 8..15 -> K+16..23;
//            lanes 16-31 same rows, K offset +8 / +24.
//   B 32x16: lanes 0-15 col=lane hold K+0..7, K+16..23; lanes 16-31 hold K+8..15, K+24..31.
//   C/D:     lane<16 -> M=r, lane>=16 -> M=r+8, N=lane&15.
// ---------------------------------------------------------------------------
template <int KT>
__global__ void gemm_bf16_wmma(const __hip_bfloat16* __restrict__ X,
                               const __hip_bfloat16* __restrict__ Wt,
                               const float* __restrict__ bias,
                               float* __restrict__ outF,
                               __hip_bfloat16* __restrict__ outB,
                               int rows, int Kout, int flags, int nNodes) {
  int wave = blockIdx.x * (blockDim.x >> 5) + (threadIdx.x >> 5);
  int lane = threadIdx.x & 31;
  int chunks = Kout >> 7;                    // 128-wide column chunks
  int total = (rows >> 4) * chunks;
  if (wave >= total) return;
  int rowTile = wave / chunks;
  int ch = wave - rowTile * chunks;

  int rA  = lane & 15;
  int kbA = (lane < 16) ? 0 : 8;             // A K-base offset per lane half
  int nB  = lane & 15;
  int kgB = (lane < 16) ? 0 : 8;             // B K-group per lane half

  const __hip_bfloat16* xrow  = X  + (size_t)(rowTile * 16 + rA) * KT + kbA;
  const __hip_bfloat16* wbase = Wt + (size_t)(ch * 128 + nB) * KT + kgB;

  auto loadA = [&](FragU& a, int k0) {
    a.q[0] = *(const u4v*)(const void*)(xrow + k0);        // K = kb+0..7
    a.q[1] = *(const u4v*)(const void*)(xrow + k0 + 16);   // K = kb+16..23
  };
  auto loadB4 = [&](FragU* b, int k0, int g) {
#pragma unroll
    for (int j = 0; j < 4; ++j) {
      int ct = g * 4 + j;
      b[j].q[0] = *(const u4v*)(const void*)(wbase + ct * 16 * KT + k0);
      b[j].q[1] = *(const u4v*)(const void*)(wbase + ct * 16 * KT + k0 + 16);
    }
  };

  v8f acc[8];
#pragma unroll
  for (int ct = 0; ct < 8; ++ct) acc[ct] = (v8f){};

  FragU a0, a1;
  FragU b0[4], b1[4];
  loadA(a0, 0);
  loadB4(b0, 0, 0);

#pragma unroll
  for (int k0 = 0; k0 < KT; k0 += 32) {
    loadB4(b1, k0, 1);                       // group-1 tiles: in flight over group-0 WMMAs
    __builtin_prefetch(xrow + k0 + 256, 0, 1);
#pragma unroll
    for (int j = 0; j < 4; ++j)
      acc[j] = __builtin_amdgcn_wmma_f32_16x16x32_bf16(false, a0.v, false, b0[j].v,
                                                       (short)0, acc[j], false, false);
    if (k0 + 32 < KT) {                      // next k-step: in flight over group-1 WMMAs
      loadA(a1, k0 + 32);
      loadB4(b0, k0 + 32, 0);
    }
#pragma unroll
    for (int j = 0; j < 4; ++j)
      acc[4 + j] = __builtin_amdgcn_wmma_f32_16x16x32_bf16(false, a0.v, false, b1[j].v,
                                                           (short)0, acc[4 + j], false, false);
    a0 = a1;                                 // rotation; dissolves under full unroll
  }

  int orow0 = rowTile * 16 + ((lane < 16) ? 0 : 8);
#pragma unroll
  for (int ct = 0; ct < 8; ++ct) {
    int ocol = ch * 128 + ct * 16 + (lane & 15);
    float bval = (flags & GF_BIAS) ? bias[ocol] : 0.f;
#pragma unroll
    for (int r = 0; r < 8; ++r) {
      float v = acc[ct][r] + bval;
      if (flags & GF_RELU) v = v > 0.f ? v : 0.f;
      size_t rr = (size_t)(orow0 + r);
      if (flags & GF_MODES) {          // proj: col -> (mode, dim); out [M,N,D]
        int m = ocol >> 7, dd = ocol & 127;
        outF[((size_t)m * nNodes + rr) * DD + dd] = v;
      } else if (flags & GF_BF16) {
        outB[rr * Kout + ocol] = __float2bfloat16(v);
      } else if (flags & GF_ADD) {
        outF[rr * Kout + ocol] += v;
      } else {
        outF[rr * Kout + ocol] = v;
      }
    }
  }
}

// W[din][dout] f32  ->  Wt[dout][din] bf16
__global__ void wtrans_kernel(const float* __restrict__ src,
                              __hip_bfloat16* __restrict__ dst,
                              int din, int dout) {
  int i = blockIdx.x * blockDim.x + threadIdx.x;
  if (i >= din * dout) return;
  int o = i % dout, d = i / dout;
  dst[(size_t)o * din + d] = __float2bfloat16(src[i]);
}

__global__ void f2bf_kernel(const float* __restrict__ in,
                            __hip_bfloat16* __restrict__ out, int n) {
  int i = blockIdx.x * blockDim.x + threadIdx.x;
  if (i < n) out[i] = __float2bfloat16(in[i]);
}

__global__ void fill_u32_kernel(unsigned* __restrict__ p, unsigned v, int n) {
  int i = blockIdx.x * blockDim.x + threadIdx.x;
  if (i < n) p[i] = v;
}

// Row LayerNorm over 128, wave32 per row, optional ReLU, bf16 out.
__global__ void ln_bf16_kernel(const float* __restrict__ x,
                               const float* __restrict__ gamma,
                               const float* __restrict__ beta,
                               __hip_bfloat16* __restrict__ out,
                               int rows, int relu) {
  int wid = blockIdx.x * (blockDim.x >> 5) + (threadIdx.x >> 5);
  int nw  = gridDim.x * (blockDim.x >> 5);
  int lane = threadIdx.x & 31;
  for (int r = wid; r < rows; r += nw) {
    f4v v = *(const f4v*)(const void*)(x + (size_t)r * DD + lane * 4);
    float s = v[0] + v[1] + v[2] + v[3];
    float s2 = v[0]*v[0] + v[1]*v[1] + v[2]*v[2] + v[3]*v[3];
    for (int m = 16; m > 0; m >>= 1) { s += __shfl_xor(s, m, 32); s2 += __shfl_xor(s2, m, 32); }
    float mean = s * (1.f / DD);
    float rstd = rsqrtf(s2 * (1.f / DD) - mean * mean + 1e-5f);
#pragma unroll
    for (int j = 0; j < 4; ++j) {
      int cidx = lane * 4 + j;
      float y = (v[j] - mean) * rstd * gamma[cidx] + beta[cidx];
      if (relu) y = y > 0.f ? y : 0.f;
      out[(size_t)r * DD + cidx] = __float2bfloat16(y);
    }
  }
}

// din=2 linear + LN + ReLU (wave per edge) for rel_pos / rel_ang paths.
__device__ inline void small_mlp_ln(float u, float v,
                                    const float* __restrict__ w,
                                    const float* __restrict__ b,
                                    const float* __restrict__ g,
                                    const float* __restrict__ be,
                                    __hip_bfloat16* __restrict__ out, int lane) {
  float t[4]; float s = 0.f, s2 = 0.f;
#pragma unroll
  for (int j = 0; j < 4; ++j) {
    int c = lane * 4 + j;
    t[j] = u * w[c] + v * w[DD + c] + b[c];
    s += t[j]; s2 += t[j] * t[j];
  }
  for (int m = 16; m > 0; m >>= 1) { s += __shfl_xor(s, m, 32); s2 += __shfl_xor(s2, m, 32); }
  float mean = s * (1.f / DD);
  float rstd = rsqrtf(s2 * (1.f / DD) - mean * mean + 1e-5f);
#pragma unroll
  for (int j = 0; j < 4; ++j) {
    int c = lane * 4 + j;
    float y = (t[j] - mean) * rstd * g[c] + be[c];
    out[c] = __float2bfloat16(y > 0.f ? y : 0.f);
  }
}

__global__ void edge_prep_kernel(const float* __restrict__ pos,
                                 const float* __restrict__ rotm,
                                 const float* __restrict__ rota,
                                 const int* __restrict__ ei,
                                 const float* w0, const float* b0, const float* g0, const float* be0,
                                 const float* w1, const float* b1, const float* g1, const float* be1,
                                 __hip_bfloat16* __restrict__ ha,
                                 __hip_bfloat16* __restrict__ hb) {
  int wid = blockIdx.x * (blockDim.x >> 5) + (threadIdx.x >> 5);
  int nw  = gridDim.x * (blockDim.x >> 5);
  int lane = threadIdx.x & 31;
  for (int e = wid; e < EE; e += nw) {
    int s = ei[e], d = ei[EE + e];
    float px = pos[((size_t)s * TT + (TT - 1)) * 2 + 0] - pos[((size_t)d * TT + (TT - 1)) * 2 + 0];
    float py = pos[((size_t)s * TT + (TT - 1)) * 2 + 1] - pos[((size_t)d * TT + (TT - 1)) * 2 + 1];
    const float* m = rotm + (size_t)d * 4;
    float rx = px * m[0] + py * m[2];   // einsum('ei,eij->ej')
    float ry = px * m[1] + py * m[3];
    float th = rota[s] - rota[d];
    small_mlp_ln(rx, ry, w0, b0, g0, be0, ha + (size_t)e * DD, lane);
    small_mlp_ln(cosf(th), sinf(th), w1, b1, g1, be1, hb + (size_t)e * DD, lane);
  }
}

// order-preserving float<->u32 for atomicMax-based segment max
__device__ inline unsigned fenc(float f) {
  unsigned u = __float_as_uint(f);
  return (u & 0x80000000u) ? ~u : (u | 0x80000000u);
}
__device__ inline float fdec(unsigned u) {
  return (u & 0x80000000u) ? __uint_as_float(u & 0x7fffffffu) : __uint_as_float(~u);
}

// wave per edge: coalesced 16B loads, per-head (4-lane) shuffle reduction.
__global__ void attn_alpha_kernel(const float* __restrict__ qn,
                                  const float* __restrict__ knf,
                                  const float* __restrict__ ke,
                                  const int* __restrict__ ei,
                                  float* __restrict__ alpha,
                                  unsigned* __restrict__ amaxU) {
  int wid = blockIdx.x * (blockDim.x >> 5) + (threadIdx.x >> 5);
  int nw  = gridDim.x * (blockDim.x >> 5);
  int lane = threadIdx.x & 31;
  for (int e = wid; e < EE; e += nw) {
    int s = ei[e], d = ei[EE + e];
    f4v q  = *(const f4v*)(const void*)(qn  + (size_t)d * DD + lane * 4);
    f4v k1 = *(const f4v*)(const void*)(knf + (size_t)s * DD + lane * 4);
    f4v k2 = *(const f4v*)(const void*)(ke  + (size_t)e * DD + lane * 4);
    float p = q[0]*(k1[0]+k2[0]) + q[1]*(k1[1]+k2[1])
            + q[2]*(k1[2]+k2[2]) + q[3]*(k1[3]+k2[3]);
    p += __shfl_xor(p, 1, 32);
    p += __shfl_xor(p, 2, 32);            // lanes 4h..4h+3 hold head-h sum
    if ((lane & 3) == 0) {
      int h = lane >> 2;
      float a = p * 0.25f;                 // 1/sqrt(dh), dh=16
      alpha[(size_t)e * HH + h] = a;
      atomicMax(amaxU + (size_t)d * HH + h, fenc(a));
    }
  }
}

__global__ void attn_ex_kernel(const int* __restrict__ ei,
                               float* __restrict__ alpha,
                               const unsigned* __restrict__ amaxU,
                               float* __restrict__ denom) {
  int idx = blockIdx.x * blockDim.x + threadIdx.x;
  if (idx >= EE * HH) return;
  int e = idx >> 3, h = idx & 7;
  int d = ei[EE + e];
  float ex = __expf(alpha[idx] - fdec(amaxU[d * HH + h]));
  alpha[idx] = ex;
  atomicAdd(denom + d * HH + h, ex);
}

__global__ void attn_msg_kernel(const int* __restrict__ ei,
                                const float* __restrict__ alpha,
                                const float* __restrict__ denom,
                                const float* __restrict__ vnf,
                                const float* __restrict__ ve,
                                float* __restrict__ agg) {
  int idx = blockIdx.x * blockDim.x + threadIdx.x;
  if (idx >= EE * DD) return;
  int e = idx >> 7, j = idx & 127, h = j >> 4;
  int s = ei[e], d = ei[EE + e];
  float w = alpha[(size_t)e * HH + h] / (denom[(size_t)d * HH + h] + 1e-16f);
  float val = (vnf[(size_t)s * DD + j] + ve[(size_t)e * DD + j]) * w;
  atomicAdd(agg + (size_t)d * DD + j, val);
}

__global__ void gate_kernel(const float* __restrict__ ihp,
                            const float* __restrict__ hhp,
                            const float* __restrict__ selfp,
                            const float* __restrict__ agg,
                            __hip_bfloat16* __restrict__ updb) {
  int i = blockIdx.x * blockDim.x + threadIdx.x;
  if (i >= NN * DD) return;
  float gte = 1.f / (1.f + __expf(-(ihp[i] + hhp[i])));
  float a = agg[i];
  updb[i] = __float2bfloat16(a + gte * (selfp[i] - a));
}

// ---------------------------------------------------------------------------
extern "C" void kernel_launch(void* const* d_in, const int* in_sizes, int n_in,
                              void* d_out, int out_size, void* d_ws, size_t ws_size,
                              hipStream_t stream) {
  (void)in_sizes; (void)n_in; (void)out_size; (void)ws_size;
  const float* x_car   = (const float*)d_in[0];
  const float* x_infra = (const float*)d_in[1];
  const float* pos     = (const float*)d_in[2];
  const float* rotm    = (const float*)d_in[3];
  const float* rota    = (const float*)d_in[4];
  const int*   ei      = (const int*)d_in[5];
  const float* prm     = (const float*)d_in[6];   // flat pytree (sorted-key) blob
  float*       out     = (float*)d_out;

  // ---- params blob offsets (jax pytree: dict keys sorted; lin={b,w}, ln={b,g})
  size_t off = 0;
  auto take = [&](size_t n) { size_t o = off; off += n; return o; };
  struct Lin { size_t b, w; }; struct LnP { size_t b, g; };
  auto takeLin = [&](int din, int dout) { Lin l; l.b = take(dout); l.w = take((size_t)din * dout); return l; };
  auto takeLn  = [&]() { LnP l; l.b = take(DD); l.g = take(DD); return l; };
  // top keys: layers < norm < proj < rel
  struct LayerP { Lin hh, ih, ke, kn, mlp1, mlp2; LnP n1, n2; Lin out, q, sf, ve, vn; } LP[4];
  for (int l = 0; l < 4; ++l) {          // keys: hh ih ke kn mlp1 mlp2 norm1 norm2 out q self ve vn
    LP[l].hh = takeLin(DD, DD); LP[l].ih = takeLin(DD, DD);
    LP[l].ke = takeLin(DD, DD); LP[l].kn = takeLin(DD, DD);
    LP[l].mlp1 = takeLin(DD, FFD); LP[l].mlp2 = takeLin(FFD, DD);
    LP[l].n1 = takeLn(); LP[l].n2 = takeLn();
    LP[l].out = takeLin(DD, DD); LP[l].q = takeLin(DD, DD); LP[l].sf = takeLin(DD, DD);
    LP[l].ve = takeLin(DD, DD); LP[l].vn = takeLin(DD, DD);
  }
  LnP normP = takeLn();
  Lin projP = takeLin(DD, MM * DD);
  // rel keys: aggr_l aggr_ln0 aggr_ln1 b0_l0 b0_l1 b0_ln b1_l0 b1_l1 b1_ln
  Lin aggrL = takeLin(DD, DD); LnP aggr0 = takeLn(); LnP aggr1 = takeLn();
  Lin b0l0 = takeLin(2, DD); Lin b0l1 = takeLin(DD, DD); LnP b0ln = takeLn();
  Lin b1l0 = takeLin(2, DD); Lin b1l1 = takeLin(DD, DD); LnP b1ln = takeLn();

  // ---- workspace carve (~1.05 GB)
  char* wsp = (char*)d_ws; size_t wcur = 0;
  auto carve = [&](size_t bytes) { void* p = wsp + wcur; wcur = (wcur + bytes + 255) & ~(size_t)255; return p; };
  __hip_bfloat16* wtblob = (__hip_bfloat16*)carve(3u << 20);
  float* xcar  = (float*)carve((size_t)NN * DD * 4);
  float* qn    = (float*)carve((size_t)NN * DD * 4);
  float* knf   = (float*)carve((size_t)NN * DD * 4);
  float* vnf   = (float*)carve((size_t)NN * DD * 4);
  float* selfp = (float*)carve((size_t)NN * DD * 4);
  float* hhp   = (float*)carve((size_t)NN * DD * 4);
  float* ihp   = (float*)carve((size_t)NN * DD * 4);
  float* agg   = (float*)carve((size_t)NN * DD * 4);
  __hip_bfloat16* agg_bf  = (__hip_bfloat16*)carve((size_t)NN * DD * 2);
  __hip_bfloat16* upd_bf  = (__hip_bfloat16*)carve((size_t)NN * DD * 2);
  __hip_bfloat16* xn_bf   = (__hip_bfloat16*)carve((size_t)NN * DD * 2);
  __hip_bfloat16* xinf_bf = (__hip_bfloat16*)carve((size_t)NN * DD * 2);
  __hip_bfloat16* hmlp_bf = (__hip_bfloat16*)carve((size_t)NN * FFD * 2);
  __hip_bfloat16* rel_bf  = (__hip_bfloat16*)carve((size_t)EE * DD * 2);
  __hip_bfloat16* e_bf1   = (__hip_bfloat16*)carve((size_t)EE * DD * 2);
  __hip_bfloat16* e_bf2   = (__hip_bfloat16*)carve((size_t)EE * DD * 2);
  float* ef1   = (float*)carve((size_t)EE * DD * 4);
  float* ef2   = (float*)carve((size_t)EE * DD * 4);
  float* alpha = (float*)carve((size_t)EE * HH * 4);
  unsigned* amaxU = (unsigned*)carve((size_t)NN * HH * 4);
  float* denom = (float*)carve((size_t)NN * HH * 4);

  // ---- weight transpose+convert to bf16 [dout][din]
  size_t wtcur = 0;
  auto prepW = [&](Lin w, int din, int dout) {
    __hip_bfloat16* dst = wtblob + wtcur; wtcur += (size_t)din * dout;
    int tot = din * dout;
    wtrans_kernel<<<(tot + 255) / 256, 256, 0, stream>>>(prm + w.w, dst, din, dout);
    return (const __hip_bfloat16*)dst;
  };
  struct LayerW { const __hip_bfloat16 *q, *kn, *ke, *vn, *ve, *sf, *ih, *hh, *out, *mlp1, *mlp2; } LW[4];
  for (int l = 0; l < 4; ++l) {
    LW[l].q = prepW(LP[l].q, DD, DD);   LW[l].kn = prepW(LP[l].kn, DD, DD);
    LW[l].ke = prepW(LP[l].ke, DD, DD); LW[l].vn = prepW(LP[l].vn, DD, DD);
    LW[l].ve = prepW(LP[l].ve, DD, DD); LW[l].sf = prepW(LP[l].sf, DD, DD);
    LW[l].ih = prepW(LP[l].ih, DD, DD); LW[l].hh = prepW(LP[l].hh, DD, DD);
    LW[l].out = prepW(LP[l].out, DD, DD);
    LW[l].mlp1 = prepW(LP[l].mlp1, DD, FFD);
    LW[l].mlp2 = prepW(LP[l].mlp2, FFD, DD);
  }
  const __hip_bfloat16* W_b0l1 = prepW(b0l1, DD, DD);
  const __hip_bfloat16* W_b1l1 = prepW(b1l1, DD, DD);
  const __hip_bfloat16* W_aggr = prepW(aggrL, DD, DD);
  const __hip_bfloat16* W_proj = prepW(projP, DD, MM * DD);

  auto gemm = [&](const __hip_bfloat16* X, const __hip_bfloat16* Wt, size_t boff,
                  float* oF, __hip_bfloat16* oB, int rows, int K, int Kout, int flags) {
    int waves = (rows >> 4) * (Kout >> 7);
    int blocks = (waves + 7) / 8;
    if (K == 128)
      gemm_bf16_wmma<128><<<blocks, 256, 0, stream>>>(X, Wt, prm + boff, oF, oB, rows, Kout, flags, NN);
    else
      gemm_bf16_wmma<512><<<blocks, 256, 0, stream>>>(X, Wt, prm + boff, oF, oB, rows, Kout, flags, NN);
  };
  auto lnrun = [&](const float* x, LnP p, __hip_bfloat16* o, int rows, int relu) {
    ln_bf16_kernel<<<2048, 256, 0, stream>>>(x, prm + p.g, prm + p.b, o, rows, relu);
  };

  // ---- inputs -> working buffers
  f2bf_kernel<<<(NN * DD + 255) / 256, 256, 0, stream>>>(x_infra, xinf_bf, NN * DD);
  hipMemcpyAsync(xcar, x_car, (size_t)NN * DD * 4, hipMemcpyDeviceToDevice, stream);

  // ---- rel embedding
  edge_prep_kernel<<<4096, 256, 0, stream>>>(pos, rotm, rota, ei,
      prm + b0l0.w, prm + b0l0.b, prm + b0ln.g, prm + b0ln.b,
      prm + b1l0.w, prm + b1l0.b, prm + b1ln.g, prm + b1ln.b, e_bf1, e_bf2);
  gemm(e_bf1, W_b0l1, b0l1.b, ef1, nullptr, EE, DD, DD, GF_BIAS);
  gemm(e_bf2, W_b1l1, b1l1.b, ef1, nullptr, EE, DD, DD, GF_BIAS | GF_ADD);
  lnrun(ef1, aggr0, e_bf1, EE, 1);                       // relu(ln(s))
  gemm(e_bf1, W_aggr, aggrL.b, ef2, nullptr, EE, DD, DD, GF_BIAS);
  lnrun(ef2, aggr1, rel_bf, EE, 0);                      // rel [E,128] bf16

  // ---- layers
  for (int l = 0; l < 4; ++l) {
    lnrun(xcar, LP[l].n1, xn_bf, NN, 0);
    gemm(xn_bf,  LW[l].q,  LP[l].q.b,  qn,    nullptr, NN, DD, DD, GF_BIAS);
    gemm(xn_bf,  LW[l].sf, LP[l].sf.b, selfp, nullptr, NN, DD, DD, GF_BIAS);
    gemm(xn_bf,  LW[l].hh, LP[l].hh.b, hhp,   nullptr, NN, DD, DD, GF_BIAS);
    gemm(xinf_bf, LW[l].kn, LP[l].kn.b, knf,  nullptr, NN, DD, DD, GF_BIAS);
    gemm(xinf_bf, LW[l].vn, LP[l].vn.b, vnf,  nullptr, NN, DD, DD, GF_BIAS);
    gemm(rel_bf, LW[l].ke, LP[l].ke.b, ef1,   nullptr, EE, DD, DD, GF_BIAS);
    gemm(rel_bf, LW[l].ve, LP[l].ve.b, ef2,   nullptr, EE, DD, DD, GF_BIAS);

    fill_u32_kernel<<<(NN * HH + 255) / 256, 256, 0, stream>>>(amaxU, 0u, NN * HH);
    fill_u32_kernel<<<(NN * HH + 255) / 256, 256, 0, stream>>>((unsigned*)denom, 0u, NN * HH);
    fill_u32_kernel<<<(NN * DD + 255) / 256, 256, 0, stream>>>((unsigned*)agg, 0u, NN * DD);

    attn_alpha_kernel<<<4096, 256, 0, stream>>>(qn, knf, ef1, ei, alpha, amaxU);
    attn_ex_kernel<<<(EE * HH + 255) / 256, 256, 0, stream>>>(ei, alpha, amaxU, denom);
    attn_msg_kernel<<<(EE * DD + 255) / 256, 256, 0, stream>>>(ei, alpha, denom, vnf, ef2, agg);

    f2bf_kernel<<<(NN * DD + 255) / 256, 256, 0, stream>>>(agg, agg_bf, NN * DD);
    gemm(agg_bf, LW[l].ih, LP[l].ih.b, ihp, nullptr, NN, DD, DD, GF_BIAS);
    gate_kernel<<<(NN * DD + 255) / 256, 256, 0, stream>>>(ihp, hhp, selfp, agg, upd_bf);
    gemm(upd_bf, LW[l].out, LP[l].out.b, xcar, nullptr, NN, DD, DD, GF_BIAS | GF_ADD);

    lnrun(xcar, LP[l].n2, xn_bf, NN, 0);
    gemm(xn_bf, LW[l].mlp1, LP[l].mlp1.b, nullptr, hmlp_bf, NN, DD, FFD, GF_BIAS | GF_RELU | GF_BF16);
    gemm(hmlp_bf, LW[l].mlp2, LP[l].mlp2.b, xcar, nullptr, NN, FFD, DD, GF_BIAS | GF_ADD);
  }

  // ---- head: ln -> proj -> [M,N,D]
  lnrun(xcar, normP, xn_bf, NN, 0);
  gemm(xn_bf, W_proj, projP.b, out, nullptr, NN, DD, MM * DD, GF_BIAS | GF_MODES);
}
